// LutLinear_34918084116785
// MI455X (gfx1250) — matrix operand order repaired and checked by reading, not compiled, not run on hardware
//
#include <hip/hip_runtime.h>
#include <hip/hip_bf16.h>

// ---------------- problem constants ----------------
#define KDIM 4096
#define NDIM 4096
#define MDIM 8192          // 4 * 2048
#define WBIT 3
// GEMM tiling (shared)
#define BM 128
#define BN 128
// fallback path (in-kernel x conversion, TDM for B only)
#define BK 32
#define KT_STEPS (KDIM / BK)
#define ASTRIDE_U 20       // 32 bf16 (16u) + 8 bf16 pad (4u) = 80B
#define BSTRIDE_U 20
#define BBUF_U (BN * BSTRIDE_U)
// TDM-everything path
#define BK2 64
#define KT2_STEPS (KDIM / BK2)
#define STR2_U 36          // 64 bf16 (32u) + 16B pad (4u) = 144B row stride
#define TBUF_U (BM * STR2_U)   // 4608 uints = 18KB per buffer (BM == BN)

typedef __attribute__((ext_vector_type(16))) __bf16 v16bf;
typedef __attribute__((ext_vector_type(8)))  float  v8f;
typedef unsigned int u32;
typedef unsigned long long u64;
typedef u32 u32x4 __attribute__((ext_vector_type(4)));
typedef u32 u32x8 __attribute__((ext_vector_type(8)));

union FragU {
    struct { uint4 lo; uint4 hi; } q;
    v16bf v;
};

// f32 pair -> packed bf16x2 (RNE), hardware cvt if available
__device__ __forceinline__ unsigned int pack2_bf16(float a, float b) {
#if __has_builtin(__builtin_amdgcn_cvt_pk_bf16_f32)
    typedef __bf16 bf16x2 __attribute__((ext_vector_type(2)));
    bf16x2 r = __builtin_amdgcn_cvt_pk_bf16_f32(a, b);
    return __builtin_bit_cast(unsigned int, r);
#else
    unsigned ua = __builtin_bit_cast(unsigned, a);
    unsigned ub = __builtin_bit_cast(unsigned, b);
    ua = (ua + 0x7FFFu + ((ua >> 16) & 1u)) >> 16;
    ub = (ub + 0x7FFFu + ((ub >> 16) & 1u)) & 0xFFFF0000u;
    return (ua & 0xFFFFu) | ub;
#endif
}

__device__ __forceinline__ void wait_tensorcnt_le2() {
#if __has_builtin(__builtin_amdgcn_s_wait_tensorcnt)
    __builtin_amdgcn_s_wait_tensorcnt(2);
#else
    asm volatile("s_wait_tensorcnt 0x2" ::: "memory");
#endif
}
__device__ __forceinline__ void wait_tensorcnt_le1() {
#if __has_builtin(__builtin_amdgcn_s_wait_tensorcnt)
    __builtin_amdgcn_s_wait_tensorcnt(1);
#else
    asm volatile("s_wait_tensorcnt 0x1" ::: "memory");
#endif
}
__device__ __forceinline__ void wait_tensorcnt_le0() {
#if __has_builtin(__builtin_amdgcn_s_wait_tensorcnt)
    __builtin_amdgcn_s_wait_tensorcnt(0);
#else
    asm volatile("s_wait_tensorcnt 0x0" ::: "memory");
#endif
}

// -------------------------------------------------------------------------
// TDM tile load: rows x tile_k bf16 tile (K contiguous in memory, row stride
// KDIM elems) into LDS with 16B pad after every row of data.
//   tile_k = 32 -> pad_interval code 3 (16 DW = 64B),  LDS stride 80B
//   tile_k = 64 -> pad_interval code 4 (32 DW = 128B), LDS stride 144B
// D# per CDNA5 ISA (2D form, VADDR2/3 = NULL).
// -------------------------------------------------------------------------
template <int TILE_K, int PAD_CODE>
__device__ __forceinline__ void tdm_issue_tile(const unsigned short* gsrc,
                                               u32 lds_off, u32 dim1) {
    u64 ga = (u64)(uintptr_t)gsrc;
    u32x4 g0;
    g0[0] = 1u;                                            // count=1, is_restore=0
    g0[1] = lds_off;                                       // lds_addr [63:32]
    g0[2] = (u32)ga;                                       // global_addr [95:64]
    g0[3] = ((u32)(ga >> 32) & 0x01FFFFFFu) | (2u << 30);  // addr [120:96] | type=2

    u32x8 g1;
    g1[0] = (1u << 16)            // data_size = 2 bytes
          | (1u << 20)            // pad_enable
          | ((u32)PAD_CODE << 22) // pad_interval
          | (3u << 25);           // pad_amount code 3 = 4 DWORDs (16B)
    g1[1] = ((u32)KDIM & 0xFFFFu) << 16;                   // tensor_dim0 lo16
    g1[2] = ((u32)KDIM >> 16) | ((dim1 & 0xFFFFu) << 16);  // dim0 hi | dim1 lo
    g1[3] = (dim1 >> 16) | ((u32)TILE_K << 16);            // dim1 hi | tile_dim0
    g1[4] = (u32)BM;                                       // tile_dim1 = 128
    g1[5] = (u32)KDIM;                                     // tensor_dim0_stride lo32
    g1[6] = 0u;
    g1[7] = 0u;

    asm volatile("tensor_load_to_lds %0, %1" :: "s"(g0), "s"(g1) : "memory");
}

// -------------------------------------------------------------------------
// Kernel 1: decode 3-bit binary-coded weights -> bf16, transposed Wt[n][k].
// -------------------------------------------------------------------------
__global__ void decode_weights_kernel(const int* __restrict__ bw,
                                      const float* __restrict__ alpha,
                                      const float* __restrict__ scale,
                                      unsigned short* __restrict__ Wt) {
    int tid = blockIdx.x * blockDim.x + threadIdx.x;
    int kb  = tid & 127;
    int n   = tid >> 7;
    int g   = kb >> 2;

    unsigned w0 = (unsigned)bw[(kb * WBIT + 0) * NDIM + n];
    unsigned w1 = (unsigned)bw[(kb * WBIT + 1) * NDIM + n];
    unsigned w2 = (unsigned)bw[(kb * WBIT + 2) * NDIM + n];
    float a0 = alpha[(g * WBIT + 0) * NDIM + n];
    float a1 = alpha[(g * WBIT + 1) * NDIM + n];
    float a2 = alpha[(g * WBIT + 2) * NDIM + n];
    float s  = scale[g * NDIM + n];

    unsigned packed[16];
    float vbuf[32];
#pragma unroll
    for (int j = 0; j < 32; ++j) {
        float t0 = ((w0 >> j) & 1u) ? a0 : -a0;
        float t1 = ((w1 >> j) & 1u) ? a1 : -a1;
        float t2 = ((w2 >> j) & 1u) ? a2 : -a2;
        vbuf[j] = s * (t0 + t1 + t2);
    }
#pragma unroll
    for (int j = 0; j < 16; ++j)
        packed[j] = pack2_bf16(vbuf[2 * j], vbuf[2 * j + 1]);

    uint4* dst = (uint4*)(Wt + (size_t)n * KDIM + (size_t)kb * 32);
#pragma unroll
    for (int q = 0; q < 4; ++q)
        dst[q] = make_uint4(packed[4 * q + 0], packed[4 * q + 1],
                            packed[4 * q + 2], packed[4 * q + 3]);
}

// -------------------------------------------------------------------------
// Kernel 1b: x fp32 -> bf16 (row-major pass-through), 8 elems/thread.
// -------------------------------------------------------------------------
__global__ void convert_x_kernel(const float* __restrict__ x,
                                 unsigned short* __restrict__ Xb) {
    size_t base = ((size_t)blockIdx.x * blockDim.x + threadIdx.x) * 8;
    const float4 f0 = *(const float4*)(x + base);
    const float4 f1 = *(const float4*)(x + base + 4);
    *(uint4*)(Xb + base) = make_uint4(pack2_bf16(f0.x, f0.y), pack2_bf16(f0.z, f0.w),
                                      pack2_bf16(f1.x, f1.y), pack2_bf16(f1.z, f1.w));
}

// -------------------------------------------------------------------------
// Kernel 2 (primary): TDM stages BOTH A and B bf16 panels (BK=64, double
// buffered). Main loop is just ds_load_b128 fragments + 16 WMMAs per step.
// -------------------------------------------------------------------------
__global__ void __launch_bounds__(256)
wmma_gemm_tdm2_kernel(const unsigned short* __restrict__ Xb,
                      const unsigned short* __restrict__ Wt,
                      float* __restrict__ out) {
    __shared__ __align__(16) u32 As_u[2 * TBUF_U];   // 2 x 18KB
    __shared__ __align__(16) u32 Bs_u[2 * TBUF_U];   // 2 x 18KB

    const int tid    = threadIdx.x;
    const int lane   = tid & 31;
    const int wid    = tid >> 5;
    const int wave_m = wid >> 2;
    const int wave_n = wid & 3;
    const int lr     = lane & 15;
    const int hi     = lane >> 4;

    const int blockM = blockIdx.y * BM;
    const int blockN = blockIdx.x * BN;

    v8f acc[4][2];
#pragma unroll
    for (int mi = 0; mi < 4; ++mi)
#pragma unroll
        for (int ni = 0; ni < 2; ++ni)
            acc[mi][ni] = (v8f){0.f, 0.f, 0.f, 0.f, 0.f, 0.f, 0.f, 0.f};

    const unsigned short* Apanel = Xb + (size_t)blockM * KDIM;
    const unsigned short* Bpanel = Wt + (size_t)blockN * KDIM;

    if (wid == 0) {
        tdm_issue_tile<BK2, 4>(Apanel, (u32)(uintptr_t)&As_u[0], (u32)MDIM);
        tdm_issue_tile<BK2, 4>(Bpanel, (u32)(uintptr_t)&Bs_u[0], (u32)NDIM);
    }

    for (int kt = 0; kt < KT2_STEPS; ++kt) {
        if (wid == 0) {
            if (kt + 1 < KT2_STEPS) {
                const int knext = (kt + 1) * BK2;
                const int nb    = (kt + 1) & 1;
                tdm_issue_tile<BK2, 4>(Apanel + knext, (u32)(uintptr_t)&As_u[nb * TBUF_U], (u32)MDIM);
                tdm_issue_tile<BK2, 4>(Bpanel + knext, (u32)(uintptr_t)&Bs_u[nb * TBUF_U], (u32)NDIM);
                wait_tensorcnt_le2();   // in-order: tile kt's pair has landed
            } else {
                wait_tensorcnt_le0();
            }
        }
        __syncthreads();

        const u32* Asb = &As_u[(kt & 1) * TBUF_U];
        const u32* Bsb = &Bs_u[(kt & 1) * TBUF_U];

#pragma unroll
        for (int kk = 0; kk < 2; ++kk) {   // two 16x16x32 K-steps per staged tile
            FragU afrag[4], bfrag[2];
            // A (16-bit 16x32): lane<16 K {0..7,16..23}; lane>=16 K {8..15,24..31}
#pragma unroll
            for (int mi = 0; mi < 4; ++mi) {
                int row = wave_m * 64 + mi * 16 + lr;
                int u   = row * STR2_U + kk * 16 + hi * 4;
                afrag[mi].q.lo = *(const uint4*)&Asb[u];
                afrag[mi].q.hi = *(const uint4*)&Asb[u + 8];
            }
            // B (16-bit 32x16): lane<16 K {0..7,16..23}; lane>=16 K {8..15,24..31}
#pragma unroll
            for (int ni = 0; ni < 2; ++ni) {
                int col = wave_n * 32 + ni * 16 + lr;
                int u   = col * STR2_U + kk * 16 + hi * 4;
                bfrag[ni].q.lo = *(const uint4*)&Bsb[u];
                bfrag[ni].q.hi = *(const uint4*)&Bsb[u + 8];
            }
#pragma unroll
            for (int mi = 0; mi < 4; ++mi)
#pragma unroll
                for (int ni = 0; ni < 2; ++ni)
                    acc[mi][ni] = __builtin_amdgcn_wmma_f32_16x16x32_bf16(
                        false, afrag[mi].v, false, bfrag[ni].v,
                        (short)0, acc[mi][ni], false, false);
        }
        __syncthreads();   // all reads of buf[kt&1] done before its next TDM
    }

#pragma unroll
    for (int mi = 0; mi < 4; ++mi) {
#pragma unroll
        for (int ni = 0; ni < 2; ++ni) {
            int col = blockN + wave_n * 32 + ni * 16 + lr;
#pragma unroll
            for (int i = 0; i < 8; ++i) {
                int row = blockM + wave_m * 64 + mi * 16 + hi * 8 + i;
                out[(size_t)row * NDIM + col] = acc[mi][ni][i];
            }
        }
    }
}

// -------------------------------------------------------------------------
// Kernel 2 (fallback, ws too small for Xb): in-kernel x conversion, TDM-B.
// -------------------------------------------------------------------------
__global__ void __launch_bounds__(256)
wmma_gemm_fallback_kernel(const float* __restrict__ x,
                          const unsigned short* __restrict__ Wt,
                          float* __restrict__ out) {
    __shared__ __align__(16) u32 As_u[BM * ASTRIDE_U];
    __shared__ __align__(16) u32 Bs_u[2 * BBUF_U];

    const int tid    = threadIdx.x;
    const int lane   = tid & 31;
    const int wid    = tid >> 5;
    const int wave_m = wid >> 2;
    const int wave_n = wid & 3;
    const int lr     = lane & 15;
    const int hi     = lane >> 4;

    const int blockM = blockIdx.y * BM;
    const int blockN = blockIdx.x * BN;

    v8f acc[4][2];
#pragma unroll
    for (int mi = 0; mi < 4; ++mi)
#pragma unroll
        for (int ni = 0; ni < 2; ++ni)
            acc[mi][ni] = (v8f){0.f, 0.f, 0.f, 0.f, 0.f, 0.f, 0.f, 0.f};

    const int a_r0 = tid >> 3;
    const int a_c4 = tid & 7;
    const unsigned short* Bpanel = Wt + (size_t)blockN * KDIM;

    if (wid == 0)
        tdm_issue_tile<BK, 3>(Bpanel, (u32)(uintptr_t)&Bs_u[0], (u32)NDIM);

    for (int kt = 0; kt < KT_STEPS; ++kt) {
        const int k0 = kt * BK;
#pragma unroll
        for (int p = 0; p < 4; ++p) {
            int row = a_r0 + p * 32;
            const float4 f = *(const float4*)(x + (size_t)(blockM + row) * KDIM + k0 + a_c4 * 4);
            int idx = row * ASTRIDE_U + a_c4 * 2;
            As_u[idx]     = pack2_bf16(f.x, f.y);
            As_u[idx + 1] = pack2_bf16(f.z, f.w);
        }
        if (wid == 0) {
            if (kt + 1 < KT_STEPS) {
                tdm_issue_tile<BK, 3>(Bpanel + (k0 + BK),
                                      (u32)(uintptr_t)&Bs_u[((kt + 1) & 1) * BBUF_U],
                                      (u32)NDIM);
                wait_tensorcnt_le1();
            } else {
                wait_tensorcnt_le0();
            }
        }
        __syncthreads();

        const u32* Bsb = &Bs_u[(kt & 1) * BBUF_U];
        FragU afrag[4], bfrag[2];
#pragma unroll
        for (int mi = 0; mi < 4; ++mi) {
            int row = wave_m * 64 + mi * 16 + lr;
            int u   = row * ASTRIDE_U + hi * 4;
            afrag[mi].q.lo = *(const uint4*)&As_u[u];
            afrag[mi].q.hi = *(const uint4*)&As_u[u + 8];
        }
#pragma unroll
        for (int ni = 0; ni < 2; ++ni) {
            int col = wave_n * 32 + ni * 16 + lr;
            int u   = col * BSTRIDE_U + hi * 4;
            bfrag[ni].q.lo = *(const uint4*)&Bsb[u];
            bfrag[ni].q.hi = *(const uint4*)&Bsb[u + 8];
        }
#pragma unroll
        for (int mi = 0; mi < 4; ++mi)
#pragma unroll
            for (int ni = 0; ni < 2; ++ni)
                acc[mi][ni] = __builtin_amdgcn_wmma_f32_16x16x32_bf16(
                    false, afrag[mi].v, false, bfrag[ni].v,
                    (short)0, acc[mi][ni], false, false);
        __syncthreads();
    }

#pragma unroll
    for (int mi = 0; mi < 4; ++mi) {
#pragma unroll
        for (int ni = 0; ni < 2; ++ni) {
            int col = blockN + wave_n * 32 + ni * 16 + lr;
#pragma unroll
            for (int i = 0; i < 8; ++i) {
                int row = blockM + wave_m * 64 + mi * 16 + hi * 8 + i;
                out[(size_t)row * NDIM + col] = acc[mi][ni][i];
            }
        }
    }
}

extern "C" void kernel_launch(void* const* d_in, const int* in_sizes, int n_in,
                              void* d_out, int out_size, void* d_ws, size_t ws_size,
                              hipStream_t stream) {
    const float* x      = (const float*)d_in[0];
    const int*   bw     = (const int*)d_in[1];
    const float* alpha  = (const float*)d_in[2];
    const float* scale  = (const float*)d_in[3];
    float*       out    = (float*)d_out;

    const size_t wt_bytes = (size_t)KDIM * NDIM * 2;   // 32 MB
    const size_t xb_bytes = (size_t)MDIM * KDIM * 2;   // 64 MB
    unsigned short* Wt = (unsigned short*)d_ws;

    decode_weights_kernel<<<(KDIM / 32) * NDIM / 256, 256, 0, stream>>>(bw, alpha, scale, Wt);

    dim3 grid(NDIM / BN, MDIM / BM);   // N fastest -> A panels stay hot in L2
    if (ws_size >= wt_bytes + xb_bytes) {
        unsigned short* Xb = (unsigned short*)((char*)d_ws + wt_bytes);
        convert_x_kernel<<<(size_t)MDIM * KDIM / (256 * 8), 256, 0, stream>>>(x, Xb);
        wmma_gemm_tdm2_kernel<<<grid, 256, 0, stream>>>(Xb, Wt, out);
    } else {
        wmma_gemm_fallback_kernel<<<grid, 256, 0, stream>>>(x, Wt, out);
    }
}
